// Discriminator_39883066310761
// MI455X (gfx1250) — matrix operand order repaired and checked by reading, not compiled, not run on hardware
//
#include <hip/hip_runtime.h>
#include <hip/hip_bf16.h>

typedef __attribute__((ext_vector_type(4)))  float  v4f;
typedef __attribute__((ext_vector_type(4)))  __bf16 v4bf;
typedef __attribute__((ext_vector_type(8)))  __bf16 v8bf;
typedef __attribute__((ext_vector_type(16))) __bf16 v16bf;
typedef __attribute__((ext_vector_type(8)))  float  v8f;

#define DIM   572
#define DP    576          // DIM padded to multiple of 32 (wmma K)
#define RNUM  6
#define SH    584          // padded LDS row stride (elements); 584*2B = 16B-aligned rows
#define WAVES 4
#define TPB   (WAVES * 32)

// LDS layout (bf16 elements):
//   [0,        32*SH)            Mt_hi : transposed M panel, 32 cols x SH
//   [32*SH,    64*SH)            Mt_lo
//   per wave w at 64*SH + w*32*SH: Ahi (16 x SH) then Alo (16 x SH)
#define MT_HI_OFF  0
#define MT_LO_OFF  (32 * SH)
#define H_BASE_OFF (64 * SH)
#define H_WAVE_SZ  (32 * SH)
#define SMEM_ELEMS (64 * SH + WAVES * 32 * SH)
#define SMEM_BYTES (SMEM_ELEMS * 2)

#define Z8 {0.f, 0.f, 0.f, 0.f, 0.f, 0.f, 0.f, 0.f}

__device__ __forceinline__ void split_bf16(float x, __bf16& hi, __bf16& lo) {
  hi = (__bf16)x;
  lo = (__bf16)(x - (float)hi);
}

__device__ __forceinline__ v16bf cat8(v8bf a, v8bf b) {
  return __builtin_shufflevector(a, b, 0, 1, 2, 3, 4, 5, 6, 7,
                                       8, 9, 10, 11, 12, 13, 14, 15);
}

__global__ void __launch_bounds__(TPB)
score_kernel(const float* __restrict__ node_emb,
             const float* __restrict__ rel,
             const float* __restrict__ fake_f,
             const int*   __restrict__ src_idx,   // [R,E]
             const int*   __restrict__ dst_idx,   // [R,E] (ignored if use_fake)
             float*       __restrict__ out,       // [R*E] section
             int E, int use_fake)
{
  extern __shared__ __bf16 lds[];
  const int tid  = threadIdx.x;
  const int wave = tid >> 5;
  const int lane = tid & 31;
  const int grp  = lane >> 4;   // 0: lanes 0-15, 1: lanes 16-31
  const int ln   = lane & 15;
  const int r    = blockIdx.y;
  const int tile_base = (blockIdx.x * WAVES + wave) * 16;

  __bf16* Mt_hi = lds + MT_HI_OFF;
  __bf16* Mt_lo = lds + MT_LO_OFF;
  __bf16* Ahi   = lds + H_BASE_OFF + wave * H_WAVE_SZ;
  __bf16* Alo   = Ahi + 16 * SH;

  // ---- gather 16 src rows (b128 loads), split to bf16 hi/lo, stage per-wave ----
  const int* srcp = src_idx + (size_t)r * E;
  for (int row = 0; row < 16; ++row) {
    int e   = tile_base + row;
    int idx = (e < E) ? srcp[e] : 0;
    const float* hrow = node_emb + (size_t)idx * DIM;
    for (int i = lane; i < DP / 4; i += 32) {         // 143 real float4s + 1 pad
      v4f x = {0.f, 0.f, 0.f, 0.f};
      if (i < DIM / 4) x = *(const v4f*)(hrow + i * 4);
      v4bf hi, lo;
#pragma unroll
      for (int q = 0; q < 4; ++q) {
        __bf16 h_, l_;
        split_bf16(x[q], h_, l_);
        hi[q] = h_; lo[q] = l_;
      }
      *(v4bf*)(Ahi + row * SH + i * 4) = hi;
      *(v4bf*)(Alo + row * SH + i * 4) = lo;
    }
  }

  // ---- per-lane row pointers for the "other" operand (accumulator layout) ----
  const float* orow[8];
  const int* dstp = dst_idx + (size_t)r * E;
#pragma unroll
  for (int v = 0; v < 8; ++v) {
    int e  = tile_base + v + 8 * grp;
    int ec = (e < E) ? e : 0;
    if (use_fake) {
      orow[v] = fake_f + ((size_t)r * E + ec) * DIM;
    } else {
      orow[v] = node_emb + (size_t)dstp[ec] * DIM;
    }
  }

  const float* relr = rel + (size_t)r * DIM * DIM;

  float acc[8];
#pragma unroll
  for (int v = 0; v < 8; ++v) acc[v] = 0.0f;

  for (int fb = 0; fb < DP / 32; ++fb) {
    // ---- cooperative fill of transposed, hi/lo-split M panel (b128 loads) ----
    __syncthreads();
    for (int j = tid; j < DP * 8; j += TPB) {       // DP rows x 8 float4s
      int dd   = j >> 3;                            // k index 0..575
      int c4   = j & 7;                             // float4 within 32-col window
      int colb = fb * 32 + c4 * 4;
      v4f x = {0.f, 0.f, 0.f, 0.f};
      if (dd < DIM && colb + 3 < DIM)               // 572 % 4 == 0: no straddle
        x = *(const v4f*)(relr + (size_t)dd * DIM + colb);
#pragma unroll
      for (int q = 0; q < 4; ++q) {
        __bf16 h_, l_;
        split_bf16(x[q], h_, l_);
        Mt_hi[(c4 * 4 + q) * SH + dd] = h_;
        Mt_lo[(c4 * 4 + q) * SH + dd] = l_;
      }
    }
    __syncthreads();

    // ---- prefetch next fb's rel panel into cache while we compute this one ----
    if (fb + 1 < DP / 32) {
      const float* np = relr + (size_t)(fb + 1) * 32;
      for (int dd = tid; dd < DIM; dd += TPB)
        __builtin_prefetch(np + (size_t)dd * DIM, 0, 1);
    }

    // 6 independent accumulator chains: one per split-bf16 product term
    v8f T0a = Z8, T0b = Z8, T0c = Z8;
    v8f T1a = Z8, T1b = Z8, T1c = Z8;

    const __bf16* arow_h = Ahi + ln * SH + grp * 16;
    const __bf16* arow_l = Alo + ln * SH + grp * 16;
    const __bf16* b0r_h  = Mt_hi + ln * SH + grp * 16;
    const __bf16* b0r_l  = Mt_lo + ln * SH + grp * 16;
    const __bf16* b1r_h  = Mt_hi + (ln + 16) * SH + grp * 16;
    const __bf16* b1r_l  = Mt_lo + (ln + 16) * SH + grp * 16;

    for (int kb = 0; kb < DP / 32; ++kb) {
      int k0 = kb * 32;
      // A operand: lane row = ln; K contiguous per lane (k = 16*grp + s)
      v16bf Ah = cat8(*(const v8bf*)(arow_h + k0), *(const v8bf*)(arow_h + k0 + 8));
      v16bf Al = cat8(*(const v8bf*)(arow_l + k0), *(const v8bf*)(arow_l + k0 + 8));
      // B operands: lane column = ln (+16 for second half); K contiguous per lane
      v16bf B0h = cat8(*(const v8bf*)(b0r_h + k0), *(const v8bf*)(b0r_h + k0 + 8));
      v16bf B0l = cat8(*(const v8bf*)(b0r_l + k0), *(const v8bf*)(b0r_l + k0 + 8));
      v16bf B1h = cat8(*(const v8bf*)(b1r_h + k0), *(const v8bf*)(b1r_h + k0 + 8));
      v16bf B1l = cat8(*(const v8bf*)(b1r_l + k0), *(const v8bf*)(b1r_l + k0 + 8));

      // split-bf16 fp32-accurate product: hi*hi + hi*lo + lo*hi
      T0a = __builtin_amdgcn_wmma_f32_16x16x32_bf16(false, Ah, false, B0h, (short)0, T0a, false, false);
      T1a = __builtin_amdgcn_wmma_f32_16x16x32_bf16(false, Ah, false, B1h, (short)0, T1a, false, false);
      T0b = __builtin_amdgcn_wmma_f32_16x16x32_bf16(false, Ah, false, B0l, (short)0, T0b, false, false);
      T1b = __builtin_amdgcn_wmma_f32_16x16x32_bf16(false, Ah, false, B1l, (short)0, T1b, false, false);
      T0c = __builtin_amdgcn_wmma_f32_16x16x32_bf16(false, Al, false, B0h, (short)0, T0c, false, false);
      T1c = __builtin_amdgcn_wmma_f32_16x16x32_bf16(false, Al, false, B1h, (short)0, T1c, false, false);
    }

    // ---- stage 2: elementwise dot with O, in accumulator layout ----
    int c0 = fb * 32 + ln;     // max 544+15 = 559 < 572, always valid
    int c1 = c0 + 16;          // may exceed DIM on last block (T1 is 0 there)
#pragma unroll
    for (int v = 0; v < 8; ++v) {
      float t0 = T0a[v] + T0b[v] + T0c[v];
      float t1 = T1a[v] + T1b[v] + T1c[v];
      float o0 = orow[v][c0];
      float o1 = (c1 < DIM) ? orow[v][c1] : 0.0f;
      acc[v] += t0 * o0 + t1 * o1;
    }
  }

  // ---- reduce over N (16 lanes per half) and store per-edge scores ----
  float* osec = out + (size_t)r * E;
#pragma unroll
  for (int v = 0; v < 8; ++v) {
    float s = acc[v];
    s += __shfl_xor(s, 1, 32);
    s += __shfl_xor(s, 2, 32);
    s += __shfl_xor(s, 4, 32);
    s += __shfl_xor(s, 8, 32);
    if (ln == v) {
      int e = tile_base + v + 8 * grp;
      if (e < E) osec[e] = s;
    }
  }
}

// ---- graph embedding tail: column-mean partials, then tiny MLP ----
__global__ void __launch_bounds__(192)
mean_partial_kernel(const float* __restrict__ node_emb,
                    float* __restrict__ part, int Nn, int slices)
{
  int col = blockIdx.x * blockDim.x + threadIdx.x;
  int sl  = blockIdx.y;
  if (col >= DIM) return;
  int rows = (Nn + slices - 1) / slices;
  int r0 = sl * rows;
  int r1 = r0 + rows; if (r1 > Nn) r1 = Nn;
  float s = 0.0f;
  for (int row = r0; row < r1; ++row) s += node_emb[(size_t)row * DIM + col];
  part[(size_t)sl * DIM + col] = s;
}

__global__ void __launch_bounds__(256)
mlp_kernel(const float* __restrict__ part, int slices, int Nn,
           const float* __restrict__ w1, const float* __restrict__ b1,
           const float* __restrict__ w2, const float* __restrict__ b2,
           float* __restrict__ outg)
{
  __shared__ float hg[DIM];
  __shared__ float h[DIM / 2];
  __shared__ float red[256];
  int tid = threadIdx.x;
  for (int c = tid; c < DIM; c += 256) {
    float s = 0.0f;
    for (int sl = 0; sl < slices; ++sl) s += part[(size_t)sl * DIM + c];
    hg[c] = s / (float)Nn;
  }
  __syncthreads();
  for (int j = tid; j < DIM / 2; j += 256) {
    float s = b1[j];
    for (int d = 0; d < DIM; ++d) s += hg[d] * w1[(size_t)d * (DIM / 2) + j];
    h[j] = (s > 0.0f) ? s : 0.0f;
  }
  __syncthreads();
  float p = 0.0f;
  for (int j = tid; j < DIM / 2; j += 256) p += h[j] * w2[j];
  red[tid] = p;
  __syncthreads();
  for (int off = 128; off > 0; off >>= 1) {
    if (tid < off) red[tid] += red[tid + off];
    __syncthreads();
  }
  if (tid == 0) outg[0] = red[0] + b2[0];
}

extern "C" void kernel_launch(void* const* d_in, const int* in_sizes, int n_in,
                              void* d_out, int out_size, void* d_ws, size_t ws_size,
                              hipStream_t stream) {
  const float* node_emb = (const float*)d_in[0];
  const float* rel      = (const float*)d_in[1];
  const float* fake_f   = (const float*)d_in[2];
  const float* w1       = (const float*)d_in[3];
  const float* b1       = (const float*)d_in[4];
  const float* w2       = (const float*)d_in[5];
  const float* b2       = (const float*)d_in[6];
  const int* pos_src    = (const int*)d_in[7];
  const int* pos_dst    = (const int*)d_in[8];
  const int* n1_src     = (const int*)d_in[9];
  const int* n1_dst     = (const int*)d_in[10];
  const int* n2_src     = (const int*)d_in[11];
  const int* n2_dst     = (const int*)d_in[12];
  float* out = (float*)d_out;

  const int E  = in_sizes[7] / RNUM;     // 30000
  const int Nn = in_sizes[0] / DIM;      // 50000

  // allow >64KB dynamic LDS (ignore error; harmless if already allowed)
  (void)hipFuncSetAttribute((const void*)score_kernel,
                            hipFuncAttributeMaxDynamicSharedMemorySize,
                            SMEM_BYTES);

  dim3 grid((E + WAVES * 16 - 1) / (WAVES * 16), RNUM);
  dim3 block(TPB);
  size_t sec = (size_t)RNUM * E;

  score_kernel<<<grid, block, SMEM_BYTES, stream>>>(node_emb, rel, fake_f,
                                                    pos_src, pos_dst, out, E, 0);
  score_kernel<<<grid, block, SMEM_BYTES, stream>>>(node_emb, rel, fake_f,
                                                    n1_src, n1_dst, out + sec, E, 0);
  score_kernel<<<grid, block, SMEM_BYTES, stream>>>(node_emb, rel, fake_f,
                                                    n2_src, n2_dst, out + 2 * sec, E, 1);

  float* part = (float*)d_ws;
  int slices = 64;
  if ((size_t)slices * DIM * sizeof(float) > ws_size) {
    slices = (int)(ws_size / (DIM * sizeof(float)));
    if (slices < 1) slices = 1;
  }
  mean_partial_kernel<<<dim3(3, slices), 192, 0, stream>>>(node_emb, part, Nn, slices);
  mlp_kernel<<<1, 256, 0, stream>>>(part, slices, Nn, w1, b1, w2, b2, out + 3 * sec);
}
